// EGNNModel_45535243272652
// MI455X (gfx1250) — matrix operand order repaired and checked by reading, compiled, not run on hardware
//
#include <hip/hip_runtime.h>
#include <hip/hip_bf16.h>
#include <math.h>
#include <cstddef>

// ---------------------------------------------------------------------------
// EGNN on MI455X (gfx1250): bf16 WMMA for edge/node MLP GEMMs, f32 accumulate.
// Edge kernel: 32 edges per wave (2 M-tiles) so every B fragment feeds 2 WMMAs.
// ---------------------------------------------------------------------------

static constexpr int kN = 10000;
static constexpr int kE = 250000;
static constexpr int kH = 128;
static constexpr int kL = 4;
static constexpr int kG = 256;
static constexpr int kF = 11;

typedef __attribute__((ext_vector_type(16))) __bf16 bf16x16;
typedef __attribute__((ext_vector_type(8)))  __bf16 bf16x8;
typedef __attribute__((ext_vector_type(8)))  float  v8f;

__device__ __forceinline__ float sigm_f(float v) { return 1.f / (1.f + __expf(-v)); }
__device__ __forceinline__ float silu_f(float v) { return v / (1.f + __expf(-v)); }

// A fragment (16-bit A 16x32 layout): lane<16 -> K {0..7,16..23}, lane>=16 -> K {8..15,24..31}
__device__ __forceinline__ bf16x16 load_a_frag(const __bf16* __restrict__ p, int koff) {
  bf16x8 lo = *(const bf16x8*)(p + koff);
  bf16x8 hi = *(const bf16x8*)(p + koff + 16);
  bf16x16 a;
#pragma unroll
  for (int q = 0; q < 8; ++q) { a[q] = lo[q]; a[q + 8] = hi[q]; }
  return a;
}

// ---------------------------------------------------------------------------
// Weight swizzle: f32 [K x 128] row-major -> bf16 WMMA B fragments.
// Fragment (nt, kt): 32 lanes x 16 halves contiguous.
//   lane 0..15  hold col = nt*16+lane,    K = kt*32 + {0..7,16..23}
//   lane 16..31 hold col = nt*16+lane-16, K = kt*32 + {8..15,24..31}
// out[((nt*KS + kt)*32 + lane)*16 + i]
// ---------------------------------------------------------------------------
__global__ void swz_kernel(const float* __restrict__ W, __bf16* __restrict__ out, int K) {
  int tid = blockIdx.x * blockDim.x + threadIdx.x;
  int total = K * 128;
  if (tid >= total) return;
  int i    = tid & 15;
  int lane = (tid >> 4) & 31;
  int rem  = tid >> 9;
  int KS   = K >> 5;
  int kt   = rem % KS;
  int nt   = rem / KS;
  int k    = kt * 32 + ((lane >> 4) << 3) + (i & 7) + ((i >> 3) << 4);
  int n    = nt * 16 + (lane & 15);
  out[tid] = (__bf16)W[(size_t)k * 128 + n];
}

// h = relu(x @ emb_w + emb_b), write f32 + bf16 copies
__global__ void embed_kernel(const float* __restrict__ x, const float* __restrict__ w,
                             const float* __restrict__ b, float* __restrict__ hf,
                             __bf16* __restrict__ hb) {
  int tid = blockIdx.x * blockDim.x + threadIdx.x;
  if (tid >= kN * kH) return;
  int n = tid >> 7, j = tid & 127;
  float s = b[j];
#pragma unroll
  for (int f = 0; f < kF; ++f) s += x[n * kF + f] * w[f * kH + j];
  s = fmaxf(s, 0.f);
  hf[tid] = s;
  hb[tid] = (__bf16)s;
}

__global__ void zero_kernel(float* __restrict__ p, int n) {
  int t = blockIdx.x * blockDim.x + threadIdx.x;
  if (t < n) p[t] = 0.f;
}

__global__ void gsum_kernel(const float* __restrict__ pos, const int* __restrict__ batch,
                            float* __restrict__ gsum) {
  int n = blockIdx.x * blockDim.x + threadIdx.x;
  if (n >= kN) return;
  int b = batch[n];
  atomicAdd(&gsum[b * 4 + 0], pos[n * 3 + 0]);
  atomicAdd(&gsum[b * 4 + 1], pos[n * 3 + 1]);
  atomicAdd(&gsum[b * 4 + 2], pos[n * 3 + 2]);
  atomicAdd(&gsum[b * 4 + 3], 1.f);
}

__global__ void posc_kernel(const float* __restrict__ pos, const int* __restrict__ batch,
                            const float* __restrict__ gsum, float* __restrict__ pc) {
  int n = blockIdx.x * blockDim.x + threadIdx.x;
  if (n >= kN) return;
  int b = batch[n];
  float c = fmaxf(gsum[b * 4 + 3], 1.f);
#pragma unroll
  for (int k = 0; k < 3; ++k) pc[n * 3 + k] = pos[n * 3 + k] - gsum[b * 4 + k] / c;
}

// ---------------------------------------------------------------------------
// Edge kernel. 8 waves x 32 edges = 256 edges/block.
//   ein = [h[dst] | h[src] | dist]  (257)
//   m = silu(silu(ein@W1+b1)@W2+b2); a = sigmoid(m@aw+ab); scatter m*a -> agg[dst]
// GEMM1 (dual M-tile, shared B): 8 K-steps, dist column folded in as VALU FMA.
// GEMM2 + attention + scatter processed per tile (wave-private LDS m1 buffer).
// ---------------------------------------------------------------------------
__global__ __launch_bounds__(256) void egnn_edge_wmma(
    const __bf16* __restrict__ hbf, const float* __restrict__ posc,
    const int* __restrict__ ei,
    const __bf16* __restrict__ w1s, const __bf16* __restrict__ w2s,
    const float* __restrict__ w1last,
    const float* __restrict__ b1, const float* __restrict__ b2,
    const float* __restrict__ aw, const float* __restrict__ abp,
    float* __restrict__ agg) {
  __shared__ int    sSrc[8][32];
  __shared__ int    sDst[8][32];
  __shared__ float  sDist[8][32];
  __shared__ float  sAtt[8][16];
  __shared__ __bf16 sM1[8][16][kH];  // per-wave m1 tile buffer (reused per M-tile)

  const int  lane = threadIdx.x & 31;
  const int  wv   = threadIdx.x >> 5;
  const long e0   = (long)blockIdx.x * 256 + wv * 32;

  {
    long e = e0 + lane;
    int s = 0, d = 0;
    float dist = 0.f;
    if (e < kE) {
      s = ei[e];
      d = ei[kE + e];
      float dx = posc[(size_t)d * 3 + 0] - posc[(size_t)s * 3 + 0];
      float dy = posc[(size_t)d * 3 + 1] - posc[(size_t)s * 3 + 1];
      float dz = posc[(size_t)d * 3 + 2] - posc[(size_t)s * 3 + 2];
      dist = sqrtf(dx * dx + dy * dy + dz * dz);
    }
    sSrc[wv][lane] = s;
    sDst[wv][lane] = d;
    sDist[wv][lane] = dist;
  }
  __syncthreads();

  const int r      = lane & 15;            // A row within tile
  const int koff   = (lane & 16) ? 8 : 0;  // K sub-group for A fragments
  const int rowoff = koff;                 // C/D row offset
  const int col0   = lane & 15;            // C/D column within N-tile

  const __bf16* rowI0 = hbf + (size_t)sDst[wv][r] * kH;        // tile0 h_i
  const __bf16* rowJ0 = hbf + (size_t)sSrc[wv][r] * kH;        // tile0 h_j
  const __bf16* rowI1 = hbf + (size_t)sDst[wv][r + 16] * kH;   // tile1 h_i
  const __bf16* rowJ1 = hbf + (size_t)sSrc[wv][r + 16] * kH;   // tile1 h_j
  __builtin_prefetch(rowJ0, 0, 0);
  __builtin_prefetch(rowJ1, 0, 0);

  // ---- GEMM1: two [16 x 256] tiles x [256 x 128], B shared ----
  v8f acc0[8] = {}, acc1[8] = {};
#pragma unroll
  for (int kt = 0; kt < 8; ++kt) {
    const __bf16* ap0 = (kt < 4) ? (rowI0 + kt * 32) : (rowJ0 + (kt - 4) * 32);
    const __bf16* ap1 = (kt < 4) ? (rowI1 + kt * 32) : (rowJ1 + (kt - 4) * 32);
    bf16x16 a0 = load_a_frag(ap0, koff);
    bf16x16 a1 = load_a_frag(ap1, koff);
#pragma unroll
    for (int nt = 0; nt < 8; ++nt) {
      bf16x16 b = *(const bf16x16*)(w1s + (((size_t)nt * 8 + kt) * 32 + lane) * 16);
      acc0[nt] = __builtin_amdgcn_wmma_f32_16x16x32_bf16(false, a0, false, b, (short)0,
                                                         acc0[nt], false, false);
      acc1[nt] = __builtin_amdgcn_wmma_f32_16x16x32_bf16(false, a1, false, b, (short)0,
                                                         acc1[nt], false, false);
    }
  }

  const float ab0 = abp[0];

  // Per-tile epilogue: dist/bias/silu -> LDS m1, GEMM2, attention gate, scatter.
  auto process_tile = [&](v8f(&acc)[8], int rbase) {
    // dist column (K=256) + bias + silu, stage m1 in wave-private LDS
#pragma unroll
    for (int nt = 0; nt < 8; ++nt) {
      const int   col = nt * 16 + col0;
      const float wd  = w1last[col];
      const float bb  = b1[col];
#pragma unroll
      for (int q = 0; q < 8; ++q) {
        const int row = rowoff + q;
        float v = acc[nt][q] + sDist[wv][rbase + row] * wd + bb;
        sM1[wv][row][col] = (__bf16)silu_f(v);
      }
    }
    // ---- GEMM2: [16 x 128] x [128 x 128] ----
    v8f acc2[8] = {};
#pragma unroll
    for (int kt = 0; kt < 4; ++kt) {
      bf16x16 a = load_a_frag(&sM1[wv][r][kt * 32], koff);
#pragma unroll
      for (int nt = 0; nt < 8; ++nt) {
        bf16x16 b = *(const bf16x16*)(w2s + (((size_t)nt * 4 + kt) * 32 + lane) * 16);
        acc2[nt] = __builtin_amdgcn_wmma_f32_16x16x32_bf16(false, a, false, b, (short)0,
                                                           acc2[nt], false, false);
      }
    }
    // bias + silu; attention partials (per-wave LDS reduction, in-order DS)
    if (lane < 16) sAtt[wv][lane] = 0.f;
    float part[8] = {};
#pragma unroll
    for (int nt = 0; nt < 8; ++nt) {
      const int   col = nt * 16 + col0;
      const float bb  = b2[col];
      const float av  = aw[col];
#pragma unroll
      for (int q = 0; q < 8; ++q) {
        float m = silu_f(acc2[nt][q] + bb);
        acc2[nt][q] = m;
        part[q] += m * av;
      }
    }
#pragma unroll
    for (int q = 0; q < 8; ++q) atomicAdd(&sAtt[wv][rowoff + q], part[q]);
    // gated scatter-add into agg[dst] (L2-resident)
#pragma unroll
    for (int q = 0; q < 8; ++q) {
      const int  row = rowoff + q;
      const long e   = e0 + rbase + row;
      if (e < kE) {
        const float gate = sigm_f(sAtt[wv][row] + ab0);
        float* dp = agg + (size_t)sDst[wv][rbase + row] * kH;
#pragma unroll
        for (int nt = 0; nt < 8; ++nt)
          atomicAdd(dp + nt * 16 + col0, acc2[nt][q] * gate);
      }
    }
  };

  process_tile(acc0, 0);
  process_tile(acc1, 16);
}

// ---------------------------------------------------------------------------
// Node kernel: upd = silu([h|agg]@W1+b1)@W2+b2 ; h' = LN(h+upd)*g+b
// ---------------------------------------------------------------------------
__global__ __launch_bounds__(256) void egnn_node_wmma(
    const __bf16* __restrict__ hbf, const float* __restrict__ hf,
    const float* __restrict__ agg,
    const __bf16* __restrict__ n1s, const __bf16* __restrict__ n2s,
    const float* __restrict__ b1, const float* __restrict__ b2,
    const float* __restrict__ lng, const float* __restrict__ lnb,
    float* __restrict__ hf_o, __bf16* __restrict__ hbf_o) {
  __shared__ int    sNode[8][16];
  __shared__ float  sSum[8][16];
  __shared__ float  sSq[8][16];
  __shared__ __bf16 sM1[8][16][kH];

  const int lane = threadIdx.x & 31;
  const int wv   = threadIdx.x >> 5;
  const int n0   = blockIdx.x * 128 + wv * 16;

  if (lane < 16) {
    int n = n0 + lane;
    sNode[wv][lane] = (n < kN) ? n : (kN - 1);
    sSum[wv][lane] = 0.f;
    sSq[wv][lane] = 0.f;
  }
  __syncthreads();

  const int r      = lane & 15;
  const int koff   = (lane & 16) ? 8 : 0;
  const int rowoff = koff;
  const int col0   = lane & 15;
  const int myNode = sNode[wv][r];

  const __bf16* hrow = hbf + (size_t)myNode * kH;
  const float*  arow = agg + (size_t)myNode * kH;

  // ---- GEMM1: [16 x 256] x [256 x 128], A = [h | agg] ----
  v8f acc[8] = {};
#pragma unroll
  for (int kt = 0; kt < 8; ++kt) {
    bf16x16 a;
    if (kt < 4) {
      a = load_a_frag(hrow + kt * 32, koff);
    } else {
      const float* p = arow + (kt - 4) * 32;
#pragma unroll
      for (int q = 0; q < 8; ++q) {
        a[q]     = (__bf16)p[koff + q];
        a[q + 8] = (__bf16)p[koff + 16 + q];
      }
    }
#pragma unroll
    for (int nt = 0; nt < 8; ++nt) {
      bf16x16 b = *(const bf16x16*)(n1s + (((size_t)nt * 8 + kt) * 32 + lane) * 16);
      acc[nt] = __builtin_amdgcn_wmma_f32_16x16x32_bf16(false, a, false, b, (short)0,
                                                        acc[nt], false, false);
    }
  }

#pragma unroll
  for (int nt = 0; nt < 8; ++nt) {
    const int   col = nt * 16 + col0;
    const float bb  = b1[col];
#pragma unroll
    for (int q = 0; q < 8; ++q)
      sM1[wv][rowoff + q][col] = (__bf16)silu_f(acc[nt][q] + bb);
  }
  __syncthreads();

  // ---- GEMM2: [16 x 128] x [128 x 128] ----
  v8f acc2[8] = {};
#pragma unroll
  for (int kt = 0; kt < 4; ++kt) {
    bf16x16 a = load_a_frag(&sM1[wv][r][kt * 32], koff);
#pragma unroll
    for (int nt = 0; nt < 8; ++nt) {
      bf16x16 b = *(const bf16x16*)(n2s + (((size_t)nt * 4 + kt) * 32 + lane) * 16);
      acc2[nt] = __builtin_amdgcn_wmma_f32_16x16x32_bf16(false, a, false, b, (short)0,
                                                         acc2[nt], false, false);
    }
  }

  // residual + LayerNorm stats
  float ps[8] = {}, ps2[8] = {};
#pragma unroll
  for (int nt = 0; nt < 8; ++nt) {
    const int   col = nt * 16 + col0;
    const float bb  = b2[col];
#pragma unroll
    for (int q = 0; q < 8; ++q) {
      const int row = rowoff + q;
      float hr = hf[(size_t)sNode[wv][row] * kH + col] + acc2[nt][q] + bb;
      acc2[nt][q] = hr;
      ps[q] += hr;
      ps2[q] += hr * hr;
    }
  }
#pragma unroll
  for (int q = 0; q < 8; ++q) {
    atomicAdd(&sSum[wv][rowoff + q], ps[q]);
    atomicAdd(&sSq[wv][rowoff + q], ps2[q]);
  }
  __syncthreads();

#pragma unroll
  for (int q = 0; q < 8; ++q) {
    const int   row = rowoff + q;
    const int   n   = n0 + row;
    const float mu  = sSum[wv][row] * (1.f / kH);
    const float var = sSq[wv][row] * (1.f / kH) - mu * mu;
    const float rs  = rsqrtf(var + 1e-5f);
    if (n < kN) {
#pragma unroll
      for (int nt = 0; nt < 8; ++nt) {
        const int col = nt * 16 + col0;
        float o = (acc2[nt][q] - mu) * rs * lng[col] + lnb[col];
        hf_o[(size_t)n * kH + col]  = o;
        hbf_o[(size_t)n * kH + col] = (__bf16)o;
      }
    }
  }
}

// ---------------------------------------------------------------------------
// Pooling + head
// ---------------------------------------------------------------------------
__global__ void pool_kernel(const float* __restrict__ h, const int* __restrict__ batch,
                            float* __restrict__ hgsum, float* __restrict__ gcnt) {
  int tid = blockIdx.x * blockDim.x + threadIdx.x;
  if (tid >= kN * kH) return;
  int n = tid >> 7, j = tid & 127;
  int b = batch[n];
  atomicAdd(&hgsum[(size_t)b * kH + j], h[tid]);
  if (j == 0) atomicAdd(&gcnt[b], 1.f);
}

__global__ void head1_kernel(const float* __restrict__ hgsum, const float* __restrict__ gcnt,
                             const float* __restrict__ w1, const float* __restrict__ b1,
                             float* __restrict__ t) {
  int tid = blockIdx.x * blockDim.x + threadIdx.x;
  if (tid >= kG * kH) return;
  int g = tid >> 7, c = tid & 127;
  float cnt = fmaxf(gcnt[g], 1.f);
  float s = b1[c];
  for (int k = 0; k < kH; ++k) s += (hgsum[(size_t)g * kH + k] / cnt) * w1[k * kH + c];
  t[tid] = fmaxf(s, 0.f);
}

__global__ void head2_kernel(const float* __restrict__ t, const float* __restrict__ w2,
                             const float* __restrict__ b2, float* __restrict__ out) {
  int g = blockIdx.x * blockDim.x + threadIdx.x;
  if (g >= kG) return;
  float s = b2[0];
  for (int c = 0; c < kH; ++c) s += t[(size_t)g * kH + c] * w2[c];
  out[g] = s;
}

// ---------------------------------------------------------------------------
extern "C" void kernel_launch(void* const* d_in, const int* in_sizes, int n_in,
                              void* d_out, int out_size, void* d_ws, size_t ws_size,
                              hipStream_t stream) {
  (void)in_sizes; (void)n_in; (void)out_size; (void)ws_size;

  const float* x       = (const float*)d_in[0];
  const float* pos     = (const float*)d_in[1];
  const int*   ei      = (const int*)d_in[2];
  const int*   batch   = (const int*)d_in[3];
  const float* emb_w   = (const float*)d_in[4];
  const float* emb_b   = (const float*)d_in[5];
  const float* edge_w1 = (const float*)d_in[6];
  const float* edge_b1 = (const float*)d_in[7];
  const float* edge_w2 = (const float*)d_in[8];
  const float* edge_b2 = (const float*)d_in[9];
  const float* attn_w  = (const float*)d_in[10];
  const float* attn_b  = (const float*)d_in[11];
  const float* node_w1 = (const float*)d_in[12];
  const float* node_b1 = (const float*)d_in[13];
  const float* node_w2 = (const float*)d_in[14];
  const float* node_b2 = (const float*)d_in[15];
  const float* ln_g    = (const float*)d_in[16];
  const float* ln_b    = (const float*)d_in[17];
  const float* out_w1  = (const float*)d_in[18];
  const float* out_b1  = (const float*)d_in[19];
  const float* out_w2  = (const float*)d_in[20];
  const float* out_b2  = (const float*)d_in[21];
  float* out = (float*)d_out;

  char* w = (char*)d_ws;
  size_t off = 0;
  auto take = [&](size_t bytes) -> char* {
    char* p = w + off;
    off += (bytes + 255) & ~(size_t)255;
    return p;
  };

  float*  hfA   = (float*)take((size_t)kN * kH * sizeof(float));
  float*  hfB   = (float*)take((size_t)kN * kH * sizeof(float));
  __bf16* hbA   = (__bf16*)take((size_t)kN * kH * sizeof(__bf16));
  __bf16* hbB   = (__bf16*)take((size_t)kN * kH * sizeof(__bf16));
  float*  agg   = (float*)take((size_t)kN * kH * sizeof(float));
  float*  posc  = (float*)take((size_t)kN * 3 * sizeof(float));
  float*  gsum  = (float*)take((size_t)kG * 4 * sizeof(float));
  float*  hgsum = (float*)take((size_t)kG * kH * sizeof(float));
  float*  gcnt  = (float*)take((size_t)kG * sizeof(float));
  float*  tbuf  = (float*)take((size_t)kG * kH * sizeof(float));
  __bf16* w1s[kL];
  __bf16* w2s[kL];
  __bf16* n1s[kL];
  __bf16* n2s[kL];
  for (int l = 0; l < kL; ++l) {
    w1s[l] = (__bf16*)take((size_t)256 * 128 * sizeof(__bf16));
    w2s[l] = (__bf16*)take((size_t)128 * 128 * sizeof(__bf16));
    n1s[l] = (__bf16*)take((size_t)256 * 128 * sizeof(__bf16));
    n2s[l] = (__bf16*)take((size_t)128 * 128 * sizeof(__bf16));
  }

  const dim3 blk(256);

  // Weight swizzles (bf16 WMMA B-fragment layout)
  for (int l = 0; l < kL; ++l) {
    swz_kernel<<<(256 * 128 + 255) / 256, blk, 0, stream>>>(edge_w1 + (size_t)l * 257 * 128, w1s[l], 256);
    swz_kernel<<<(128 * 128 + 255) / 256, blk, 0, stream>>>(edge_w2 + (size_t)l * 128 * 128, w2s[l], 128);
    swz_kernel<<<(256 * 128 + 255) / 256, blk, 0, stream>>>(node_w1 + (size_t)l * 256 * 128, n1s[l], 256);
    swz_kernel<<<(128 * 128 + 255) / 256, blk, 0, stream>>>(node_w2 + (size_t)l * 128 * 128, n2s[l], 128);
  }

  embed_kernel<<<(kN * kH + 255) / 256, blk, 0, stream>>>(x, emb_w, emb_b, hfA, hbA);

  zero_kernel<<<(kG * 4 + 255) / 256, blk, 0, stream>>>(gsum, kG * 4);
  gsum_kernel<<<(kN + 255) / 256, blk, 0, stream>>>(pos, batch, gsum);
  posc_kernel<<<(kN + 255) / 256, blk, 0, stream>>>(pos, batch, gsum, posc);

  float*  hf_c = hfA;
  float*  hf_n = hfB;
  __bf16* hb_c = hbA;
  __bf16* hb_n = hbB;

  for (int l = 0; l < kL; ++l) {
    zero_kernel<<<(kN * kH + 255) / 256, blk, 0, stream>>>(agg, kN * kH);
    egnn_edge_wmma<<<(kE + 255) / 256, blk, 0, stream>>>(
        hb_c, posc, ei, w1s[l], w2s[l],
        edge_w1 + (size_t)l * 257 * 128 + 256 * 128,  // W1 row 256 (dist column)
        edge_b1 + l * kH, edge_b2 + l * kH,
        attn_w + l * kH, attn_b + l, agg);
    egnn_node_wmma<<<(kN + 127) / 128, blk, 0, stream>>>(
        hb_c, hf_c, agg, n1s[l], n2s[l],
        node_b1 + l * kH, node_b2 + l * kH,
        ln_g + l * kH, ln_b + l * kH, hf_n, hb_n);
    { float* t = hf_c; hf_c = hf_n; hf_n = t; }
    { __bf16* t = hb_c; hb_c = hb_n; hb_n = t; }
  }

  zero_kernel<<<(kG * kH + 255) / 256, blk, 0, stream>>>(hgsum, kG * kH);
  zero_kernel<<<(kG + 255) / 256, blk, 0, stream>>>(gcnt, kG);
  pool_kernel<<<(kN * kH + 255) / 256, blk, 0, stream>>>(hf_c, batch, hgsum, gcnt);
  head1_kernel<<<(kG * kH + 255) / 256, blk, 0, stream>>>(hgsum, gcnt, out_w1, out_b1, tbuf);
  head2_kernel<<<1, 256, 0, stream>>>(tbuf, out_w2, out_b2, out);
}
